// MultiHeadedAttention_48086453846304
// MI455X (gfx1250) — matrix operand order repaired and checked by reading, not compiled, not run on hardware
//
#include <hip/hip_runtime.h>
#include <hip/hip_bf16.h>
#include <math.h>

#define B_  4
#define S_  1024
#define H_  1024
#define HN_ 16
#define D_  64

typedef __bf16 bf16;
typedef __attribute__((ext_vector_type(16))) __bf16 v16bf;
typedef __attribute__((ext_vector_type(8)))  __bf16 v8bf;
typedef __attribute__((ext_vector_type(4)))  __bf16 v4bf;
typedef __attribute__((ext_vector_type(8)))  float  v8f;

// ---------------------------------------------------------------------------
// Fragment loaders matching CDNA5 WMMA bf16 VGPR layouts (ISA 7.12.2).
// A (16x32, MxK): lane = {half = lane>>4, m = lane&15}; element e maps to
//   k = e + 8*half (e<8),  k = (e-8) + 16 + 8*half (e>=8)
// -> two contiguous 8-element (16B) groups at offsets 8*half and 8*half+16.
static __device__ __forceinline__ v16bf load_a_frag(const bf16* p) {
    v8bf lo = *(const v8bf*)(p);
    v8bf hi = *(const v8bf*)(p + 16);
    v16bf r;
#pragma unroll
    for (int i = 0; i < 8; ++i) { r[i] = lo[i]; r[i + 8] = hi[i]; }
    return r;
}
// B (32x16, KxN) with the N x K source row-major (computing A*B^T):
// lane = {half, n = lane&15}; same k order as A: two 16B loads at 8*half, 8*half+16.
static __device__ __forceinline__ v16bf load_bt_frag(const bf16* p) {
    return load_a_frag(p);
}

static __device__ __forceinline__ v8f wmma_bf16(v16bf a, v16bf b, v8f c) {
    return __builtin_amdgcn_wmma_f32_16x16x32_bf16(
        false, a, false, b, (short)0, c, false, false);
}

// CDNA5 async global->LDS copy (16B per lane), tracked by ASYNCcnt.
// LDS offset = low 32 bits of the generic pointer (LDS aperture: low word is
// the offset from the workgroup LDS base, which is what VDST wants).
static __device__ __forceinline__ void async_cp16(unsigned lds_off,
                                                  const void* gaddr) {
    asm volatile("global_load_async_to_lds_b128 %0, %1, off"
                 :: "v"(lds_off), "v"((unsigned long long)(uintptr_t)gaddr)
                 : "memory");
}
static __device__ __forceinline__ void wait_async0() {
    asm volatile("s_wait_asynccnt 0" ::: "memory");
}
static __device__ __forceinline__ unsigned lds_off_of(const void* p) {
    return (unsigned)(uintptr_t)p;
}

// ---------------------------------------------------------------------------
// Kernel 1: QKV projection.  X[4096,1024]f32 * W[1024,1024]f32 + bias ->
// bf16, scattered head-major. vmode=0: [B*HN,S,D]; vmode=1 (V): [B*HN,D,S].
__global__ __launch_bounds__(128) void mha_proj_qkv(
    const float* __restrict__ X, const float* __restrict__ W,
    const float* __restrict__ bias, bf16* __restrict__ Out, int vmode)
{
    __shared__ bf16 ldsA[64 * 32];   // [m][k]
    __shared__ bf16 ldsB[64 * 32];   // [n][k]  (W tile transposed)
    const int tid  = threadIdx.x;
    const int lane = tid & 31, wave = tid >> 5;
    const int half = lane >> 4, l16 = lane & 15;
    const int m0 = blockIdx.y * 64;
    const int n0 = blockIdx.x * 64;

    v8f acc[4] = {};
    for (int k0 = 0; k0 < H_; k0 += 32) {
        {   // stage X tile 64x32: 16 contiguous f32 -> 2 packed b128 LDS stores
            int idx = tid * 16, m = idx >> 5, k = idx & 31;
            const float* src = X + (size_t)(m0 + m) * H_ + k0 + k;
            v8bf p0, p1;
#pragma unroll
            for (int i = 0; i < 8; ++i) { p0[i] = (bf16)src[i]; p1[i] = (bf16)src[8 + i]; }
            *(v8bf*)&ldsA[m * 32 + k]     = p0;
            *(v8bf*)&ldsA[m * 32 + k + 8] = p1;
        }
        {   // stage W tile 32x64 -> ldsB[n][k] (transposed: scalar stores)
            int idx = tid * 16, k = idx >> 6, n = idx & 63;
            const float* src = W + (size_t)(k0 + k) * (HN_ * D_) + n0 + n;
#pragma unroll
            for (int i = 0; i < 16; ++i) ldsB[(n + i) * 32 + k] = (bf16)src[i];
        }
        __syncthreads();
        v16bf a = load_a_frag(&ldsA[(wave * 16 + l16) * 32 + 8 * half]);
#pragma unroll
        for (int j = 0; j < 4; ++j) {
            v16bf b = load_bt_frag(&ldsB[(j * 16 + l16) * 32 + 8 * half]);
            acc[j] = wmma_bf16(a, b, acc[j]);
        }
        __syncthreads();
    }
#pragma unroll
    for (int j = 0; j < 4; ++j)
#pragma unroll
        for (int r = 0; r < 8; ++r) {
            int m = m0 + wave * 16 + r + 8 * half;
            int n = n0 + j * 16 + l16;
            float v = acc[j][r] + bias[n];
            int bb = m >> 10, s = m & (S_ - 1);
            int h  = n >> 6,  d = n & (D_ - 1);
            size_t o = vmode ? (((size_t)(bb * HN_ + h) * D_ + d) * S_ + s)
                             : (((size_t)(bb * HN_ + h) * S_ + s) * D_ + d);
            Out[o] = (bf16)v;
        }
}

// ---------------------------------------------------------------------------
// Kernel 2: scores = Q*K^T with K tile async-staged in LDS (shared by all
// 4 waves), fused epilogue (+pos_bias)/8 + mask + prev -> f32 prev_attn_out.
__global__ __launch_bounds__(128) void mha_scores(
    const bf16* __restrict__ Q, const bf16* __restrict__ K,
    const float* __restrict__ pbias, const float* __restrict__ mask,
    const float* __restrict__ prev, float* __restrict__ prev_out)
{
    __shared__ bf16 ldsK[64 * 64];   // [n][k], full D per tile (8 KB)
    const int tid  = threadIdx.x;
    const int lane = tid & 31, wave = tid >> 5;
    const int half = lane >> 4, l16 = lane & 15;
    const int bh = blockIdx.z, bb = bh / HN_, h = bh % HN_;
    const int m0 = blockIdx.y * 64 + wave * 16;
    const int n0 = blockIdx.x * 64;
    const bf16* Qh = Q + (size_t)bh * S_ * D_;
    const bf16* Kh = K + (size_t)bh * S_ * D_;

    {   // async-stage K rows n0..n0+63 (128B each): 64B per thread
        int n = tid >> 1;
        int byte = (tid & 1) * 64;
        const char* g = (const char*)(Kh + (size_t)(n0 + n) * D_) + byte;
        unsigned l = lds_off_of(&ldsK[n * 64]) + byte;
#pragma unroll
        for (int i = 0; i < 4; ++i) async_cp16(l + 16 * i, g + 16 * i);
    }
    wait_async0();
    __syncthreads();

    v8f acc[4] = {};
#pragma unroll
    for (int k0 = 0; k0 < D_; k0 += 32) {
        v16bf a = load_a_frag(Qh + (size_t)(m0 + l16) * D_ + k0 + 8 * half);
#pragma unroll
        for (int j = 0; j < 4; ++j) {
            v16bf b = load_bt_frag(&ldsK[(j * 16 + l16) * 64 + k0 + 8 * half]);
            acc[j] = wmma_bf16(a, b, acc[j]);
        }
    }
#pragma unroll
    for (int j = 0; j < 4; ++j)
#pragma unroll
        for (int r = 0; r < 8; ++r) {
            int m = m0 + r + 8 * half;
            int n = n0 + j * 16 + l16;
            size_t sidx = ((size_t)bh * S_ + m) * S_ + n;
            float sc = acc[j][r];
            sc = (sc + pbias[((size_t)h * S_ + m) * S_ + n]) * 0.125f; // 1/sqrt(64)
            sc += mask[((size_t)bb * S_ + m) * S_ + n];
            sc += prev[sidx];
            prev_out[sidx] = sc;
        }
}

// ---------------------------------------------------------------------------
// Kernel 3: row softmax over S=1024, f32 in -> bf16 probs.
__global__ __launch_bounds__(256) void mha_softmax(
    const float* __restrict__ scores, bf16* __restrict__ probs)
{
    __shared__ float redm[8];
    __shared__ float reds[8];
    const size_t row = blockIdx.x;
    const float* src = scores + row * S_;
    bf16* dst = probs + row * S_;
    const int tid = threadIdx.x, lane = tid & 31, wid = tid >> 5;

    float4 v = *(const float4*)(src + tid * 4);
    float mx = fmaxf(fmaxf(v.x, v.y), fmaxf(v.z, v.w));
#pragma unroll
    for (int off = 16; off; off >>= 1) mx = fmaxf(mx, __shfl_xor(mx, off, 32));
    if (lane == 0) redm[wid] = mx;
    __syncthreads();
    float rmax = redm[0];
#pragma unroll
    for (int i = 1; i < 8; ++i) rmax = fmaxf(rmax, redm[i]);

    float e0 = __expf(v.x - rmax), e1 = __expf(v.y - rmax);
    float e2 = __expf(v.z - rmax), e3 = __expf(v.w - rmax);
    float s = e0 + e1 + e2 + e3;
#pragma unroll
    for (int off = 16; off; off >>= 1) s += __shfl_xor(s, off, 32);
    if (lane == 0) reds[wid] = s;
    __syncthreads();
    float rsum = 0.f;
#pragma unroll
    for (int i = 0; i < 8; ++i) rsum += reds[i];
    float inv = 1.0f / rsum;

    v4bf o;
    o[0] = (bf16)(e0 * inv); o[1] = (bf16)(e1 * inv);
    o[2] = (bf16)(e2 * inv); o[3] = (bf16)(e3 * inv);
    *(v4bf*)(dst + tid * 4) = o;
}

// ---------------------------------------------------------------------------
// Kernel 4: ctx = P[S,S] * V.  V stored transposed [D,S]; each 64x128 chunk
// of V^T is async-staged once into LDS and reused by all 4 waves across four
// WMMA K-steps.  Output scattered to [B, S, HN*D] bf16.
__global__ __launch_bounds__(128) void mha_ctx(
    const bf16* __restrict__ P, const bf16* __restrict__ Vt,
    bf16* __restrict__ Ctx)
{
    __shared__ bf16 ldsV[64 * 128];  // [d][k-chunk] 16 KB
    const int tid  = threadIdx.x;
    const int lane = tid & 31, wave = tid >> 5;
    const int half = lane >> 4, l16 = lane & 15;
    const int bh = blockIdx.z, bb = bh / HN_, h = bh % HN_;
    const int m0 = blockIdx.y * 64 + wave * 16;
    const bf16* Ph = P  + (size_t)bh * S_ * S_;
    const bf16* Vh = Vt + (size_t)bh * D_ * S_;

    v8f acc[4] = {};
    for (int k0 = 0; k0 < S_; k0 += 128) {
        {   // async-stage V^T rows d=0..63, cols k0..k0+127 (256B/row)
            int d = tid >> 1;
            int byte = (tid & 1) * 128;
            const char* g = (const char*)(Vh + (size_t)d * S_ + k0) + byte;
            unsigned l = lds_off_of(&ldsV[d * 128]) + byte;
#pragma unroll
            for (int i = 0; i < 8; ++i) async_cp16(l + 16 * i, g + 16 * i);
        }
        wait_async0();
        __syncthreads();
#pragma unroll
        for (int kk = 0; kk < 128; kk += 32) {
            v16bf a = load_a_frag(Ph + (size_t)(m0 + l16) * S_ + k0 + kk + 8 * half);
#pragma unroll
            for (int j = 0; j < 4; ++j) {
                v16bf b = load_bt_frag(&ldsV[(j * 16 + l16) * 128 + kk + 8 * half]);
                acc[j] = wmma_bf16(a, b, acc[j]);
            }
        }
        __syncthreads();
    }
#pragma unroll
    for (int j = 0; j < 4; ++j)
#pragma unroll
        for (int r = 0; r < 8; ++r) {
            int m = m0 + r + 8 * half;       // sequence position
            int d = j * 16 + l16;            // head dim
            Ctx[((size_t)(bb * S_ + m)) * (HN_ * D_) + h * D_ + d] = (bf16)acc[j][r];
        }
}

// ---------------------------------------------------------------------------
// Kernel 5: output = Ctx[4096,1024]bf16 * Wo[1024,1024]f32 + bo -> f32.
__global__ __launch_bounds__(128) void mha_out_proj(
    const bf16* __restrict__ Ctx, const float* __restrict__ Wo,
    const float* __restrict__ bo, float* __restrict__ Out)
{
    __shared__ bf16 ldsB[64 * 32];   // Wo tile transposed [n][k]
    const int tid  = threadIdx.x;
    const int lane = tid & 31, wave = tid >> 5;
    const int half = lane >> 4, l16 = lane & 15;
    const int m0 = blockIdx.y * 64;
    const int n0 = blockIdx.x * 64;

    v8f acc[4] = {};
    for (int k0 = 0; k0 < HN_ * D_; k0 += 32) {
        {
            int idx = tid * 16, k = idx >> 6, n = idx & 63;
            const float* src = Wo + (size_t)(k0 + k) * H_ + n0 + n;
#pragma unroll
            for (int i = 0; i < 16; ++i) ldsB[(n + i) * 32 + k] = (bf16)src[i];
        }
        __syncthreads();
        v16bf a = load_a_frag(Ctx + (size_t)(m0 + wave * 16 + l16) * (HN_ * D_) + k0 + 8 * half);
#pragma unroll
        for (int j = 0; j < 4; ++j) {
            v16bf b = load_bt_frag(&ldsB[(j * 16 + l16) * 32 + 8 * half]);
            acc[j] = wmma_bf16(a, b, acc[j]);
        }
        __syncthreads();
    }
#pragma unroll
    for (int j = 0; j < 4; ++j)
#pragma unroll
        for (int r = 0; r < 8; ++r) {
            int m = m0 + wave * 16 + r + 8 * half;
            int n = n0 + j * 16 + l16;
            Out[(size_t)m * H_ + n] = acc[j][r] + bo[n];
        }
}

// ---------------------------------------------------------------------------
extern "C" void kernel_launch(void* const* d_in, const int* in_sizes, int n_in,
                              void* d_out, int out_size, void* d_ws, size_t ws_size,
                              hipStream_t stream) {
    const float* key   = (const float*)d_in[0];
    const float* value = (const float*)d_in[1];
    const float* query = (const float*)d_in[2];
    const float* mask  = (const float*)d_in[3];
    const float* pbias = (const float*)d_in[4];
    const float* prev  = (const float*)d_in[5];
    const float* Wq = (const float*)d_in[6];  const float* bq = (const float*)d_in[7];
    const float* Wk = (const float*)d_in[8];  const float* bk = (const float*)d_in[9];
    const float* Wv = (const float*)d_in[10]; const float* bv = (const float*)d_in[11];
    const float* Wo = (const float*)d_in[12]; const float* bo = (const float*)d_in[13];

    float* out_proj = (float*)d_out;                                    // [B,S,H]
    float* prev_out = out_proj + (size_t)B_ * S_ * H_;                  // [B,HN,S,S]

    bf16* q_ws  = (bf16*)d_ws;                                          // [B*HN,S,D]
    bf16* k_ws  = q_ws  + (size_t)B_ * S_ * H_;                         // [B*HN,S,D]
    bf16* v_ws  = k_ws  + (size_t)B_ * S_ * H_;                         // [B*HN,D,S]
    bf16* probs = v_ws  + (size_t)B_ * S_ * H_;                         // [B*HN,S,S]
    bf16* ctx   = probs + (size_t)B_ * HN_ * S_ * S_;                   // [B,S,HN*D]

    dim3 blk(128);
    dim3 gProj(H_ / 64, (B_ * S_) / 64);                                // 16 x 64
    mha_proj_qkv<<<gProj, blk, 0, stream>>>(query, Wq, bq, q_ws, 0);
    mha_proj_qkv<<<gProj, blk, 0, stream>>>(key,   Wk, bk, k_ws, 0);
    mha_proj_qkv<<<gProj, blk, 0, stream>>>(value, Wv, bv, v_ws, 1);

    dim3 gScore(S_ / 64, S_ / 64, B_ * HN_);                            // 16 x 16 x 64
    mha_scores<<<gScore, blk, 0, stream>>>(q_ws, k_ws, pbias, mask, prev, prev_out);

    mha_softmax<<<dim3(B_ * HN_ * S_), dim3(256), 0, stream>>>(prev_out, probs);

    dim3 gCtx(1, S_ / 64, B_ * HN_);                                    // 1 x 16 x 64
    mha_ctx<<<gCtx, blk, 0, stream>>>(probs, v_ws, ctx);

    mha_out_proj<<<gProj, blk, 0, stream>>>(ctx, Wo, bo, out_proj);
}
